// TransformerBlockCached_33268816674852
// MI455X (gfx1250) — compile-verified, matrix operand order checked by reading, not run on hardware
//
#include <hip/hip_runtime.h>

#define HIDDEN   2880
#define HEAD_DIM 64
#define N_HEADS  64
#define N_KV     8
#define N_NEW    256
#define N_CACHED 1792
#define N_TOTAL  2048
#define QKV_DIM  5120
#define QDIM     4096     // N_HEADS * HEAD_DIM
#define KVROW    512      // N_KV * HEAD_DIM
#define INTER    2880
#define NKEYS    144      // SLIDING_WINDOW + 16

typedef __attribute__((ext_vector_type(16))) __bf16 v16bf;
typedef __attribute__((ext_vector_type(8)))  __bf16 v8bf;
typedef __attribute__((ext_vector_type(8)))  float  v8f;
typedef __attribute__((ext_vector_type(4)))  float  v4f;

// ---- WMMA fragment loaders (CDNA5 16-bit 16x32 A layout; B = W^T so lane n
// reads row n of W with the identical K pattern). p = row_base + kbase + hi,
// where hi = 8 for lanes 16..31. Elements [0..7] = K+0..7, [8..15] = K+16..23.
__device__ __forceinline__ v16bf frag_bf16(const __bf16* p) {
  v8bf lo = *reinterpret_cast<const v8bf*>(p);
  v8bf hi = *reinterpret_cast<const v8bf*>(p + 16);
  v16bf r;
#pragma unroll
  for (int e = 0; e < 8; ++e) { r[e] = lo[e]; r[8 + e] = hi[e]; }
  return r;
}

__device__ __forceinline__ v16bf frag_f32(const float* p) {
  v4f a0 = *reinterpret_cast<const v4f*>(p);
  v4f a1 = *reinterpret_cast<const v4f*>(p + 4);
  v4f b0 = *reinterpret_cast<const v4f*>(p + 16);
  v4f b1 = *reinterpret_cast<const v4f*>(p + 20);
  v16bf r;
#pragma unroll
  for (int e = 0; e < 4; ++e) {
    r[e]      = (__bf16)a0[e];
    r[4 + e]  = (__bf16)a1[e];
    r[8 + e]  = (__bf16)b0[e];
    r[12 + e] = (__bf16)b1[e];
  }
  return r;
}

__device__ __forceinline__ v8f wmma_bf16(v16bf a, v16bf b, v8f c) {
  return __builtin_amdgcn_wmma_f32_16x16x32_bf16(false, a, false, b, (short)0, c,
                                                 false, false);
}

// gfx1250 async global->LDS copy (ASYNCcnt-tracked, GVS addressing mode:
// SGPR64 base + 32-bit per-lane byte offset). 16 bytes per lane.
__device__ __forceinline__ void async_b128_to_lds(unsigned int lds_off,
                                                  int goff,
                                                  const void* gbase) {
  asm volatile("global_load_async_to_lds_b128 %0, %1, %2"
               :
               : "v"(lds_off), "v"(goff), "s"(gbase)
               : "memory");
}

__device__ __forceinline__ void wait_async0() {
  asm volatile("s_wait_asynccnt 0x0" ::: "memory");
}

// ---------------------------------------------------------------------------
// C[256 x N] = A_bf16[256 x lda(:K)] * W_f32[N x K]^T (+bias)(+residual)
// or, if wexp != nullptr:  C[row, n] += wexp[row*8] * (dot + bias[n])
// Block: 128 threads = 4 waves; wave w covers cols [bx*64 + w*16, +16).
// grid.y = 2 -> each block holds 8 M-tiles (128 rows) of acc in registers so
// each fp32 weight fragment is loaded once and reused by 8 WMMAs. All A
// fragments for a k-step are fetched into registers *before* the WMMA batch
// so the b128 loads clause together and the 8 WMMAs issue back-to-back.
// ---------------------------------------------------------------------------
__global__ __launch_bounds__(128) void gemm_k(
    const __bf16* __restrict__ A, int lda,
    const float* __restrict__ W,
    const float* __restrict__ bias,
    const float* __restrict__ residual,
    const float* __restrict__ wexp,
    float* __restrict__ C, int N, int K) {
  const int lane  = threadIdx.x & 31;
  const int wave  = threadIdx.x >> 5;
  const int n0    = blockIdx.x * 64 + wave * 16;
  const int m0    = blockIdx.y * 128;
  const int nlane = n0 + (lane & 15);
  const int hi    = (lane & 16) ? 8 : 0;
  const float* wrow = W + (size_t)nlane * K;
  const __bf16* arow0 = A + (size_t)(m0 + (lane & 15)) * lda + hi;

  v8f zero = {0.f, 0.f, 0.f, 0.f, 0.f, 0.f, 0.f, 0.f};
  v8f acc[8];
#pragma unroll
  for (int mt = 0; mt < 8; ++mt) acc[mt] = zero;

  for (int k = 0; k < K; k += 32) {
    // keep HBM weight stream deep: prefetch ~8 k-steps (1KB) ahead
    __builtin_prefetch(wrow + k + 256, 0, 3);
    v16bf b = frag_f32(wrow + k + hi);
    v16bf a[8];
#pragma unroll
    for (int mt = 0; mt < 8; ++mt)
      a[mt] = frag_bf16(arow0 + (size_t)(mt * 16) * lda + k);
#pragma unroll
    for (int mt = 0; mt < 8; ++mt)
      acc[mt] = wmma_bf16(a[mt], b, acc[mt]);
  }

  const float bv    = bias ? bias[nlane] : 0.f;
  const int   rbase = (lane & 16) ? 8 : 0;
#pragma unroll
  for (int mt = 0; mt < 8; ++mt) {
#pragma unroll
    for (int r = 0; r < 8; ++r) {
      const int    row = m0 + mt * 16 + rbase + r;
      const size_t idx = (size_t)row * N + nlane;
      float v = acc[mt][r] + bv;
      if (residual) v += residual[idx];
      if (wexp) C[idx] += wexp[row * 8] * v;
      else      C[idx] = v;
    }
  }
}

// ---------------------------------------------------------------------------
__global__ __launch_bounds__(256) void rmsnorm_k(const float* __restrict__ x,
                                                 const float* __restrict__ scale,
                                                 __bf16* __restrict__ out) {
  const int m = blockIdx.x;
  const float* row = x + (size_t)m * HIDDEN;
  float ss = 0.f;
  for (int j = threadIdx.x; j < HIDDEN; j += 256) { float v = row[j]; ss += v * v; }
#pragma unroll
  for (int off = 16; off; off >>= 1) ss += __shfl_xor(ss, off);
  __shared__ float red[8];
  const int lane = threadIdx.x & 31, wv = threadIdx.x >> 5;
  if (lane == 0) red[wv] = ss;
  __syncthreads();
  if (wv == 0) {
    float t = (lane < 8) ? red[lane] : 0.f;
#pragma unroll
    for (int off = 4; off; off >>= 1) t += __shfl_xor(t, off);
    if (lane == 0) red[0] = t;
  }
  __syncthreads();
  const float inv = rsqrtf(red[0] / (float)HIDDEN + 1e-5f);
  for (int j = threadIdx.x; j < HIDDEN; j += 256)
    out[(size_t)m * HIDDEN + j] = (__bf16)(row[j] * inv * scale[j]);
}

// ---------------------------------------------------------------------------
__global__ __launch_bounds__(256) void rope_k(const float* __restrict__ qkv,
                                              __bf16* __restrict__ qb,
                                              float* __restrict__ Kout,
                                              float* __restrict__ Vout) {
  const int m = blockIdx.x;
  const float* row = qkv + (size_t)m * QKV_DIM;
  const int p = N_CACHED + m;
  for (int t = threadIdx.x; t < 72 * 32; t += 256) {
    const int h = t >> 5;        // 0..63 q heads, 64..71 k heads
    const int d = t & 31;
    const float inv_freq = __expf(-((float)d / 32.f) * __logf(150000.f));
    float s, c;
    sincosf((float)p * inv_freq, &s, &c);
    if (h < 64) {
      const float x1 = row[h * 64 + d], x2 = row[h * 64 + d + 32];
      qb[(size_t)m * QDIM + h * 64 + d]      = (__bf16)(x1 * c - x2 * s);
      qb[(size_t)m * QDIM + h * 64 + d + 32] = (__bf16)(x2 * c + x1 * s);
    } else {
      const int kvh = h - 64;
      const float x1 = row[QDIM + kvh * 64 + d], x2 = row[QDIM + kvh * 64 + d + 32];
      float* kr = Kout + (size_t)(N_CACHED + m) * KVROW + kvh * 64;
      kr[d]      = x1 * c - x2 * s;
      kr[d + 32] = x2 * c + x1 * s;
    }
  }
  for (int t = threadIdx.x; t < KVROW; t += 256)
    Vout[(size_t)(N_CACHED + m) * KVROW + t] = row[QDIM + KVROW + t];
}

// ---------------------------------------------------------------------------
// sliding-window attention with sink: one block per (q head, 16-query tile).
// V window is DMA'd into LDS with async-to-LDS loads that overlap the S=QK^T
// WMMA work; S softmax (+sink) in LDS; P*V in VALU (tiny FLOP share).
// ---------------------------------------------------------------------------
__global__ __launch_bounds__(256) void attn_k(const __bf16* __restrict__ qb,
                                              const float* __restrict__ Kc,
                                              const float* __restrict__ Vc,
                                              const float* __restrict__ sinks,
                                              __bf16* __restrict__ attn) {
  const int h      = blockIdx.x;       // q head
  const int i0     = blockIdx.y * 16;  // query tile base
  const int kvh    = h >> 3;
  const int jstart = N_CACHED + i0 - 128;

  __shared__ float S[16][160];
  __shared__ float Vl[NKEYS * 64];
  __shared__ float red[16][16];

  // async DMA of the V window into LDS (9 x b128 per thread, ASYNCcnt-tracked)
  for (int t = threadIdx.x; t < NKEYS * 16; t += 256) {
    const int j = t >> 4, q4 = t & 15;
    const unsigned int lds_off =
        (unsigned int)(uintptr_t)(&Vl[j * 64 + q4 * 4]);
    const int goff =
        (int)(((size_t)(jstart + j) * KVROW + kvh * 64 + q4 * 4) * sizeof(float));
    async_b128_to_lds(lds_off, goff, (const void*)Vc);
  }

  const int lane = threadIdx.x & 31, wave = threadIdx.x >> 5;
  const int hi = (lane & 16) ? 8 : 0;
  v8f zero = {0.f, 0.f, 0.f, 0.f, 0.f, 0.f, 0.f, 0.f};
  for (int t = wave; t < 9; t += 8) {
    v8f acc = zero;
    const int j = jstart + t * 16 + (lane & 15);
    const float* krow = Kc + (size_t)j * KVROW + kvh * 64;
#pragma unroll
    for (int kk = 0; kk < 64; kk += 32) {
      v16bf b = frag_f32(krow + kk + hi);
      v16bf a = frag_bf16(qb + (size_t)(i0 + (lane & 15)) * QDIM + h * 64 + kk + hi);
      acc = wmma_bf16(a, b, acc);
    }
    const int rbase = (lane & 16) ? 8 : 0;
    const int cc = t * 16 + (lane & 15);
#pragma unroll
    for (int r = 0; r < 8; ++r) {
      const int qi = rbase + r;
      const int pos = N_CACHED + i0 + qi;
      float v = acc[r] * 0.125f;  // 1/sqrt(64)
      if (j > pos || j < pos - 128) v = -1e30f;
      S[qi][cc] = v;
    }
  }
  wait_async0();   // V window landed in LDS
  __syncthreads();

  const int row = threadIdx.x >> 4, c = threadIdx.x & 15;
  float mx = -1e30f;
  for (int jj = c; jj < NKEYS; jj += 16) mx = fmaxf(mx, S[row][jj]);
  red[row][c] = mx;
  __syncthreads();
  if (c == 0) {
    float m2 = sinks[h];
    for (int q = 0; q < 16; ++q) m2 = fmaxf(m2, red[row][q]);
    red[row][0] = m2;
  }
  __syncthreads();
  mx = red[row][0];
  float sum = 0.f;
  for (int jj = c; jj < NKEYS; jj += 16) {
    float e = __expf(S[row][jj] - mx);
    S[row][jj] = e;
    sum += e;
  }
  __syncthreads();
  red[row][c] = sum;
  __syncthreads();
  if (c == 0) {
    float s2 = __expf(sinks[h] - mx);
    for (int q = 0; q < 16; ++q) s2 += red[row][q];
    red[row][0] = s2;
  }
  __syncthreads();
  const float rinv = 1.f / red[row][0];
  for (int jj = c; jj < NKEYS; jj += 16) S[row][jj] *= rinv;
  __syncthreads();

  const int d = threadIdx.x & 63;
  const int rr = threadIdx.x >> 6;
#pragma unroll
  for (int rb = 0; rb < 4; ++rb) {
    const int r = rb * 4 + rr;
    float a = 0.f;
    for (int jj = 0; jj < NKEYS; ++jj) a += S[r][jj] * Vl[jj * 64 + d];
    attn[(size_t)(i0 + r) * QDIM + h * 64 + d] = (__bf16)a;
  }
}

// ---------------------------------------------------------------------------
__global__ __launch_bounds__(256) void gate_k(const __bf16* __restrict__ t2,
                                              const float* __restrict__ gw,
                                              const float* __restrict__ gb,
                                              float* __restrict__ we) {
  const int m = blockIdx.x;
  const int lane = threadIdx.x & 31, wv = threadIdx.x >> 5;  // wv = expert
  const __bf16* trow = t2 + (size_t)m * HIDDEN;
  const float* wrow = gw + wv * HIDDEN;
  float s = 0.f;
  for (int j = lane; j < HIDDEN; j += 32) s += (float)trow[j] * wrow[j];
#pragma unroll
  for (int off = 16; off; off >>= 1) s += __shfl_xor(s, off);
  __shared__ float g[8];
  if (lane == 0) g[wv] = s + gb[wv];
  __syncthreads();
  if (threadIdx.x == 0) {
    int i0 = 0;
    for (int e = 1; e < 8; ++e) if (g[e] > g[i0]) i0 = e;
    int i1 = -1;
    for (int e = 0; e < 8; ++e) {
      if (e == i0) continue;
      if (i1 < 0 || g[e] > g[i1]) i1 = e;
    }
    const float e1 = __expf(g[i1] - g[i0]);
    const float w0 = 1.f / (1.f + e1);
    for (int e = 0; e < 8; ++e) we[m * 8 + e] = 0.f;
    we[m * 8 + i0] = w0;
    we[m * 8 + i1] = e1 * w0;
  }
}

__global__ void swiglu_k(const float* __restrict__ c1, __bf16* __restrict__ h) {
  const int idx = blockIdx.x * 256 + threadIdx.x;  // over 256*2880
  const int m = idx / INTER, i = idx % INTER;
  float xg = c1[(size_t)m * (2 * INTER) + 2 * i];
  float xl = c1[(size_t)m * (2 * INTER) + 2 * i + 1];
  xg = fminf(xg, 7.f);
  xl = fminf(fmaxf(xl, -7.f), 7.f);
  const float sig = 1.f / (1.f + __expf(-1.702f * xg));
  h[idx] = (__bf16)(xg * sig * (xl + 1.f));
}

__global__ void zero_k(float* p, int n) {
  const int i = blockIdx.x * 256 + threadIdx.x;
  if (i < n) p[i] = 0.f;
}

__global__ void add_k(const float* a, const float* b, float* o, int n) {
  const int i = blockIdx.x * 256 + threadIdx.x;
  if (i < n) o[i] = a[i] + b[i];
}

// ---------------------------------------------------------------------------
extern "C" void kernel_launch(void* const* d_in, const int* in_sizes, int n_in,
                              void* d_out, int out_size, void* d_ws, size_t ws_size,
                              hipStream_t stream) {
  (void)in_sizes; (void)n_in; (void)out_size; (void)ws_size;
  const float* x        = (const float*)d_in[0];
  const float* cached_k = (const float*)d_in[1];
  const float* cached_v = (const float*)d_in[2];
  const float* n1s      = (const float*)d_in[3];
  const float* qkv_w    = (const float*)d_in[4];
  const float* qkv_b    = (const float*)d_in[5];
  const float* sinks    = (const float*)d_in[6];
  const float* out_w    = (const float*)d_in[7];
  const float* out_b    = (const float*)d_in[8];
  const float* n2s      = (const float*)d_in[9];
  const float* gate_w   = (const float*)d_in[10];
  const float* gate_b   = (const float*)d_in[11];
  const float* mlp1_w   = (const float*)d_in[12];
  const float* mlp1_b   = (const float*)d_in[13];
  const float* mlp2_w   = (const float*)d_in[14];
  const float* mlp2_b   = (const float*)d_in[15];

  float* outx = (float*)d_out;                       // [256 x 2880]
  float* Kout = outx + (size_t)N_NEW * HIDDEN;       // [2048 x 512]
  float* Vout = Kout + (size_t)N_TOTAL * KVROW;      // [2048 x 512]

  char* w = (char*)d_ws;
  __bf16* t1   = (__bf16*)w; w += (size_t)N_NEW * HIDDEN * 2;
  float*  qkv  = (float*)w;  w += (size_t)N_NEW * QKV_DIM * 4;
  __bf16* qb   = (__bf16*)w; w += (size_t)N_NEW * QDIM * 2;
  __bf16* attn = (__bf16*)w; w += (size_t)N_NEW * QDIM * 2;
  float*  x1   = (float*)w;  w += (size_t)N_NEW * HIDDEN * 4;
  __bf16* t2   = (__bf16*)w; w += (size_t)N_NEW * HIDDEN * 2;
  float*  we   = (float*)w;  w += (size_t)N_NEW * 8 * 4;
  float*  c1   = (float*)w;  w += (size_t)N_NEW * 2 * INTER * 4;
  __bf16* hb   = (__bf16*)w; w += (size_t)N_NEW * INTER * 2;
  float*  macc = (float*)w;  w += (size_t)N_NEW * HIDDEN * 4;

  hipMemcpyAsync(Kout, cached_k, (size_t)N_CACHED * KVROW * 4,
                 hipMemcpyDeviceToDevice, stream);
  hipMemcpyAsync(Vout, cached_v, (size_t)N_CACHED * KVROW * 4,
                 hipMemcpyDeviceToDevice, stream);

  rmsnorm_k<<<N_NEW, 256, 0, stream>>>(x, n1s, t1);
  gemm_k<<<dim3(QKV_DIM / 64, 2), 128, 0, stream>>>(t1, HIDDEN, qkv_w, qkv_b,
                                                    nullptr, nullptr, qkv,
                                                    QKV_DIM, HIDDEN);
  rope_k<<<N_NEW, 256, 0, stream>>>(qkv, qb, Kout, Vout);
  attn_k<<<dim3(N_HEADS, N_NEW / 16), 256, 0, stream>>>(qb, Kout, Vout, sinks, attn);
  gemm_k<<<dim3(HIDDEN / 64, 2), 128, 0, stream>>>(attn, QDIM, out_w, out_b, x,
                                                   nullptr, x1, HIDDEN, QDIM);
  rmsnorm_k<<<N_NEW, 256, 0, stream>>>(x1, n2s, t2);
  gate_k<<<N_NEW, 256, 0, stream>>>(t2, gate_w, gate_b, we);
  zero_k<<<(N_NEW * HIDDEN) / 256, 256, 0, stream>>>(macc, N_NEW * HIDDEN);

  for (int e = 0; e < 8; ++e) {
    gemm_k<<<dim3(2 * INTER / 64, 2), 128, 0, stream>>>(
        t2, HIDDEN, mlp1_w + (size_t)e * 2 * INTER * HIDDEN, mlp1_b + e * 2 * INTER,
        nullptr, nullptr, c1, 2 * INTER, HIDDEN);
    swiglu_k<<<(N_NEW * INTER) / 256, 256, 0, stream>>>(c1, hb);
    gemm_k<<<dim3(HIDDEN / 64, 2), 128, 0, stream>>>(
        hb, INTER, mlp2_w + (size_t)e * HIDDEN * INTER, mlp2_b + e * HIDDEN,
        nullptr, we + e, macc, HIDDEN, INTER);
  }
  add_k<<<(N_NEW * HIDDEN) / 256, 256, 0, stream>>>(x1, macc, outx, N_NEW * HIDDEN);
}
